// Sketch_RNN_91036126806749
// MI455X (gfx1250) — matrix operand order repaired and checked by reading, not compiled, not run on hardware
//
#include <hip/hip_runtime.h>
#include <cstdint>
#include <cstddef>

// ---------------------------------------------------------------------------
// Sketch-RNN forward pass for MI455X (gfx1250), wave32 + WMMA f16->f32.
// B=512 T=256 IN=5 H=256 Z=128 DH=512 K=20 OUT=123
// ---------------------------------------------------------------------------

typedef _Float16 v16h __attribute__((ext_vector_type(16)));
typedef _Float16 v8h  __attribute__((ext_vector_type(8)));
typedef float    v8f  __attribute__((ext_vector_type(8)));

static constexpr int BB   = 512;
static constexpr int TT   = 256;
static constexpr int NIN  = 5;
static constexpr int HEN  = 256;   // encoder hidden
static constexpr int ZZ   = 128;   // latent
static constexpr int DHH  = 512;   // decoder hidden
static constexpr int NOUT = 123;   // 6*20+3

#define DEVFN __device__ __forceinline__

DEVFN float sigf(float x) { return 1.0f / (1.0f + expf(-x)); }

// ---------------------------------------------------------------------------
// utility kernels
// ---------------------------------------------------------------------------
__global__ void k_zero_u32(uint32_t* __restrict__ p, int n) {
  int i = blockIdx.x * blockDim.x + threadIdx.x;
  if (i < n) p[i] = 0u;
}

// dst[r*cols+c] = (f16) src[r*srstride + scol + c]
__global__ void k_cvt2d_f32_f16(const float* __restrict__ src, int srstride, int scol,
                                _Float16* __restrict__ dst, int rows, int cols) {
  int i = blockIdx.x * blockDim.x + threadIdx.x;
  if (i >= rows * cols) return;
  int r = i / cols, c = i - r * cols;
  dst[i] = (_Float16)src[(size_t)r * srstride + scol + c];
}

// ---------------------------------------------------------------------------
// Fused WMMA GEMM:  out[m, coloff+n] = act( sum_k A16[m,k]*B16[n,k]
//                                           + bias1[n] + bias2[n]
//                                           + sum_{j<5} x[m,j]*Wxs[n,j]
//                                           + addsrc[m,n] )
// A16: [M,lda] f16 row-major (M = 512 = 4 blocks of 128)
// B16: [Npad,ldb] f16 row-major (weight [N,K] layout == WMMA B operand)
// Block: 256 threads = 8 waves; wave tile 32x64; WG tile 128x128.
// grid = (M/128, Npad/128)
// ---------------------------------------------------------------------------
__global__ __launch_bounds__(256)
void k_wmma_gemm(const _Float16* __restrict__ A, int lda,
                 const _Float16* __restrict__ Bw, int ldb,
                 const float* __restrict__ bias1, const float* __restrict__ bias2,
                 const float* __restrict__ addsrc, int add_rstride,
                 const float* __restrict__ xptr, int x_rstride,
                 const float* __restrict__ wxs, int wxs_rstride,
                 float* __restrict__ outf, _Float16* __restrict__ outh,
                 long out_rstride, long out_coloff,
                 int K, int Nvalid, int act) {
  const int lane  = threadIdx.x & 31;
  const int wid   = threadIdx.x >> 5;
  const int wm    = wid & 3;          // 4 waves along M
  const int wn    = wid >> 2;         // 2 waves along N
  const int m0    = blockIdx.x * 128 + wm * 32;
  const int n0    = blockIdx.y * 128 + wn * 64;
  const int r16   = lane & 15;
  const int lhalf = lane >> 4;

  v8f acc[2][4];
  v8f zacc;
#pragma unroll
  for (int j = 0; j < 8; ++j) zacc[j] = 0.0f;
#pragma unroll
  for (int mi = 0; mi < 2; ++mi)
#pragma unroll
    for (int ni = 0; ni < 4; ++ni) acc[mi][ni] = zacc;

  for (int kk = 0; kk < K; kk += 32) {
    v16h a[2], b[4];
#pragma unroll
    for (int mi = 0; mi < 2; ++mi) {
      const int m = m0 + 16 * mi + r16;
      const _Float16* p = A + (size_t)m * lda + kk + lhalf * 8;
      v8h lo = *(const v8h*)(p);
      v8h hi = *(const v8h*)(p + 16);
      a[mi] = __builtin_shufflevector(lo, hi, 0,1,2,3,4,5,6,7,8,9,10,11,12,13,14,15);
    }
    const int kb = kk + lhalf * 8;
#pragma unroll
    for (int ni = 0; ni < 4; ++ni) {
      const int n = n0 + 16 * ni + r16;
      const _Float16* p = Bw + (size_t)n * ldb + kb;
      v8h lo = *(const v8h*)(p);
      v8h hi = *(const v8h*)(p + 16);
      b[ni] = __builtin_shufflevector(lo, hi, 0,1,2,3,4,5,6,7,8,9,10,11,12,13,14,15);
    }
    if (kk + 32 < K) {  // gfx1250 global_prefetch_b8 into L1/L2
      __builtin_prefetch(A + (size_t)(m0 + r16) * lda + kk + 32, 0, 3);
      __builtin_prefetch(Bw + (size_t)(n0 + r16) * ldb + kk + 32, 0, 3);
    }
#pragma unroll
    for (int mi = 0; mi < 2; ++mi)
#pragma unroll
      for (int ni = 0; ni < 4; ++ni)
        acc[mi][ni] = __builtin_amdgcn_wmma_f32_16x16x32_f16(
            false, a[mi], false, b[ni], (short)0, acc[mi][ni], false, false);
  }

  // epilogue -----------------------------------------------------------------
  float bsum[4];
  float w5[4][5];
#pragma unroll
  for (int ni = 0; ni < 4; ++ni) {
    const int n = n0 + 16 * ni + r16;
    float bv = 0.0f;
    if (bias1) bv += bias1[n];
    if (bias2) bv += bias2[n];
    bsum[ni] = bv;
    if (wxs) {
#pragma unroll
      for (int j = 0; j < 5; ++j) w5[ni][j] = wxs[(size_t)n * wxs_rstride + j];
    }
  }
#pragma unroll
  for (int mi = 0; mi < 2; ++mi) {
#pragma unroll
    for (int r = 0; r < 8; ++r) {
      const int m = m0 + 16 * mi + lhalf * 8 + r;
      float xrow[5];
      if (wxs) {
        if (xptr) {
#pragma unroll
          for (int j = 0; j < 5; ++j) xrow[j] = xptr[(size_t)m * x_rstride + j];
        } else {  // decoder t==0: s0 = (0,0,1,0,0)
          xrow[0] = 0.f; xrow[1] = 0.f; xrow[2] = 1.f; xrow[3] = 0.f; xrow[4] = 0.f;
        }
      }
#pragma unroll
      for (int ni = 0; ni < 4; ++ni) {
        const int n = n0 + 16 * ni + r16;
        float v = acc[mi][ni][r] + bsum[ni];
        if (wxs) {
#pragma unroll
          for (int j = 0; j < 5; ++j) v += xrow[j] * w5[ni][j];
        }
        if (addsrc) v += addsrc[(size_t)m * add_rstride + n];
        if (act == 1) v = tanhf(v);
        if (n < Nvalid) {
          const size_t o = (size_t)m * out_rstride + out_coloff + n;
          if (outf) outf[o] = v;
          else      outh[o] = (_Float16)v;
        }
      }
    }
  }
}

// ---------------------------------------------------------------------------
// LSTM cell: gates G [B,4H] (PyTorch order i,f,g,o), c fp32, h written as f16
// ---------------------------------------------------------------------------
__global__ void k_lstm_cell(const float* __restrict__ G, float* __restrict__ c,
                            _Float16* __restrict__ h, int hsz, int total,
                            _Float16* __restrict__ h2, int ldh2) {
  int i = blockIdx.x * blockDim.x + threadIdx.x;
  if (i >= total) return;
  int b = i / hsz, j = i - b * hsz;
  const float* g = G + (size_t)b * 4 * hsz;
  float gi = g[j], gf = g[hsz + j], gg = g[2 * hsz + j], go = g[3 * hsz + j];
  float cc = sigf(gf) * c[i] + sigf(gi) * tanhf(gg);
  c[i] = cc;
  float hh = sigf(go) * tanhf(cc);
  h[(size_t)b * hsz + j] = (_Float16)hh;
  if (h2) h2[(size_t)b * ldh2 + j] = (_Float16)hh;
}

// z = mu + exp(presig/2) * eps
__global__ void k_latent_z(const float* __restrict__ mu, const float* __restrict__ ps,
                           const float* __restrict__ eps, _Float16* __restrict__ z, int total) {
  int i = blockIdx.x * blockDim.x + threadIdx.x;
  if (i >= total) return;
  int j = i & (ZZ - 1);
  z[i] = (_Float16)(mu[i] + expf(ps[i] * 0.5f) * eps[j]);
}

// In-place MDN transform on y[b,t,123]
__global__ void k_mdn(float* __restrict__ y, int total) {
  int i = blockIdx.x * blockDim.x + threadIdx.x;
  if (i >= total) return;
  float* p = y + (size_t)i * NOUT;
  float mx = p[0];
#pragma unroll
  for (int k = 1; k < 20; ++k) mx = fmaxf(mx, p[k * 6]);
  float e[20]; float ssum = 0.0f;
#pragma unroll
  for (int k = 0; k < 20; ++k) { e[k] = expf(p[k * 6] - mx); ssum += e[k]; }
  float inv = 1.0f / ssum;
#pragma unroll
  for (int k = 0; k < 20; ++k) {
    p[k * 6 + 0] = e[k] * inv;
    p[k * 6 + 3] = expf(p[k * 6 + 3]);
    p[k * 6 + 4] = expf(p[k * 6 + 4]);
    p[k * 6 + 5] = tanhf(p[k * 6 + 5]);
  }
  float q0 = p[120], q1 = p[121], q2 = p[122];
  float m2 = fmaxf(q0, fmaxf(q1, q2));
  float f0 = expf(q0 - m2), f1 = expf(q1 - m2), f2 = expf(q2 - m2);
  float is = 1.0f / (f0 + f1 + f2);
  p[120] = f0 * is; p[121] = f1 * is; p[122] = f2 * is;
}

// ---------------------------------------------------------------------------
extern "C" void kernel_launch(void* const* d_in, const int* in_sizes, int n_in,
                              void* d_out, int out_size, void* d_ws, size_t ws_size,
                              hipStream_t stream) {
  const float* s       = (const float*)d_in[0];
  const float* eps     = (const float*)d_in[1];
  const float* Wih_f   = (const float*)d_in[2];
  const float* Whh_f   = (const float*)d_in[3];
  const float* bih_f   = (const float*)d_in[4];
  const float* bhh_f   = (const float*)d_in[5];
  const float* Wih_b   = (const float*)d_in[6];
  const float* Whh_b   = (const float*)d_in[7];
  const float* bih_b   = (const float*)d_in[8];
  const float* bhh_b   = (const float*)d_in[9];
  const float* W_sigma = (const float*)d_in[10];
  const float* b_sigma = (const float*)d_in[11];
  const float* W_mu    = (const float*)d_in[12];
  const float* b_mu    = (const float*)d_in[13];
  const float* W_h0    = (const float*)d_in[14];
  const float* b_h0    = (const float*)d_in[15];
  const float* Wih_d   = (const float*)d_in[16];
  const float* Whh_d   = (const float*)d_in[17];
  const float* bih_d   = (const float*)d_in[18];
  const float* bhh_d   = (const float*)d_in[19];
  const float* W_dec   = (const float*)d_in[20];
  const float* b_dec   = (const float*)d_in[21];

  float* out   = (float*)d_out;                     // [B,T,123]
  float* muOut = out + (size_t)BB * TT * NOUT;      // [B,128]
  float* psOut = muOut + (size_t)BB * ZZ;           // [B,128]

  // workspace carve-up (~15.6 MB total)
  char* wp = (char*)d_ws;
  auto alloc = [&](size_t bytes) -> char* {
    char* r = wp; wp += (bytes + 255) & ~size_t(255); return r;
  };
  _Float16* whhF16 = (_Float16*)alloc((size_t)4 * HEN * HEN * 2);      // [1024,256]
  _Float16* whhB16 = (_Float16*)alloc((size_t)4 * HEN * HEN * 2);
  _Float16* whhD16 = (_Float16*)alloc((size_t)4 * DHH * DHH * 2);      // [2048,512]
  _Float16* wsig16 = (_Float16*)alloc((size_t)ZZ * 2 * HEN * 2);       // [128,512]
  _Float16* wmu16  = (_Float16*)alloc((size_t)ZZ * 2 * HEN * 2);
  _Float16* wh016  = (_Float16*)alloc((size_t)DHH * ZZ * 2);           // [512,128]
  _Float16* wz16   = (_Float16*)alloc((size_t)4 * DHH * ZZ * 2);       // [2048,128]
  _Float16* wdec16 = (_Float16*)alloc((size_t)128 * DHH * 2);          // [128(pad),512]
  _Float16* henc16 = (_Float16*)alloc((size_t)BB * 2 * HEN * 2);       // [512,512]
  _Float16* hA16   = (_Float16*)alloc((size_t)BB * HEN * 2);           // [512,256]
  float*    cF     = (float*)   alloc((size_t)BB * HEN * 4);
  float*    Gbuf   = (float*)   alloc((size_t)BB * 4 * DHH * 4);       // [512,2048]
  _Float16* z16    = (_Float16*)alloc((size_t)BB * ZZ * 2);
  _Float16* hD16   = (_Float16*)alloc((size_t)BB * DHH * 2);
  float*    cD     = (float*)   alloc((size_t)BB * DHH * 4);
  float*    Gxz    = (float*)   alloc((size_t)BB * 4 * DHH * 4);       // [512,2048]

  auto zero = [&](void* ptr, size_t bytes) {
    int n = (int)(bytes / 4);
    k_zero_u32<<<(n + 255) / 256, 256, 0, stream>>>((uint32_t*)ptr, n);
  };
  auto cvt = [&](const float* src, int srstride, int scol, _Float16* dst, int rows, int cols) {
    int n = rows * cols;
    k_cvt2d_f32_f16<<<(n + 255) / 256, 256, 0, stream>>>(src, srstride, scol, dst, rows, cols);
  };
  auto gemm = [&](const _Float16* A, int lda, const _Float16* Bw, int ldb,
                  const float* b1, const float* b2,
                  const float* add, int addstride,
                  const float* xp, const float* wxs, int wxs_stride,
                  float* of, _Float16* oh, long ors, long oco,
                  int Kd, int Nvalid, int Ntiles, int act) {
    dim3 g(BB / 128, Ntiles), blk(256);
    k_wmma_gemm<<<g, blk, 0, stream>>>(A, lda, Bw, ldb, b1, b2, add, addstride,
                                       xp, TT * NIN, wxs, wxs_stride,
                                       of, oh, ors, oco, Kd, Nvalid, act);
  };
  auto cell = [&](const float* G, float* c, _Float16* h, int hsz,
                  _Float16* h2, int ldh2) {
    int total = BB * hsz;
    k_lstm_cell<<<(total + 255) / 256, 256, 0, stream>>>(G, c, h, hsz, total, h2, ldh2);
  };

  // ---- one-time weight conversion f32 -> f16 (L2-resident, ~4MB) ----------
  cvt(Whh_f, HEN, 0, whhF16, 4 * HEN, HEN);
  cvt(Whh_b, HEN, 0, whhB16, 4 * HEN, HEN);
  cvt(Whh_d, DHH, 0, whhD16, 4 * DHH, DHH);
  cvt(W_sigma, 2 * HEN, 0, wsig16, ZZ, 2 * HEN);
  cvt(W_mu,    2 * HEN, 0, wmu16,  ZZ, 2 * HEN);
  cvt(W_h0, ZZ, 0, wh016, DHH, ZZ);
  cvt(Wih_d, NIN + ZZ, NIN, wz16, 4 * DHH, ZZ);          // z-columns of Wih_d
  zero(wdec16, (size_t)128 * DHH * 2);                   // pad rows 123..127 = 0
  cvt(W_dec, DHH, 0, wdec16, NOUT, DHH);

  // ---- forward encoder scan (full T steps) --------------------------------
  zero(hA16, (size_t)BB * HEN * 2);
  zero(cF,   (size_t)BB * HEN * 4);
  for (int t = 0; t < TT; ++t) {
    gemm(hA16, HEN, whhF16, HEN, bih_f, bhh_f, nullptr, 0,
         s + (size_t)t * NIN, Wih_f, NIN,
         Gbuf, nullptr, 4 * HEN, 0, HEN, 4 * HEN, (4 * HEN) / 128, 0);
    cell(Gbuf, cF, hA16, HEN, (t == TT - 1) ? henc16 : nullptr, 2 * HEN);
  }

  // ---- backward encoder: scan(reverse=True)[-1] == ONE step on x[T-1] -----
  zero(hA16, (size_t)BB * HEN * 2);
  zero(cF,   (size_t)BB * HEN * 4);
  gemm(hA16, HEN, whhB16, HEN, bih_b, bhh_b, nullptr, 0,
       s + (size_t)(TT - 1) * NIN, Wih_b, NIN,
       Gbuf, nullptr, 4 * HEN, 0, HEN, 4 * HEN, (4 * HEN) / 128, 0);
  cell(Gbuf, cF, hA16, HEN, henc16 + HEN, 2 * HEN);

  // ---- latent --------------------------------------------------------------
  gemm(henc16, 2 * HEN, wmu16, 2 * HEN, b_mu, nullptr, nullptr, 0, nullptr, nullptr, 0,
       muOut, nullptr, ZZ, 0, 2 * HEN, ZZ, 1, 0);
  gemm(henc16, 2 * HEN, wsig16, 2 * HEN, b_sigma, nullptr, nullptr, 0, nullptr, nullptr, 0,
       psOut, nullptr, ZZ, 0, 2 * HEN, ZZ, 1, 0);
  {
    int total = BB * ZZ;
    k_latent_z<<<(total + 255) / 256, 256, 0, stream>>>(muOut, psOut, eps, z16, total);
  }

  // ---- decoder init: h0 = tanh(z @ W_h0.T + b_h0), c0 = 0 ------------------
  gemm(z16, ZZ, wh016, ZZ, b_h0, nullptr, nullptr, 0, nullptr, nullptr, 0,
       nullptr, hD16, DHH, 0, ZZ, DHH, DHH / 128, /*tanh*/1);
  zero(cD, (size_t)BB * DHH * 4);

  // ---- hoist t-invariant decoder input term: Gxz = z @ Wz.T + bih_d+bhh_d --
  gemm(z16, ZZ, wz16, ZZ, bih_d, bhh_d, nullptr, 0, nullptr, nullptr, 0,
       Gxz, nullptr, 4 * DHH, 0, ZZ, 4 * DHH, (4 * DHH) / 128, 0);

  // ---- decoder scan + MDN head --------------------------------------------
  for (int t = 0; t < TT; ++t) {
    const float* xp = (t == 0) ? nullptr : (s + (size_t)(t - 1) * NIN);  // dec_in[t]
    gemm(hD16, DHH, whhD16, DHH, nullptr, nullptr, Gxz, 4 * DHH,
         xp, Wih_d, NIN + ZZ,
         Gbuf, nullptr, 4 * DHH, 0, DHH, 4 * DHH, (4 * DHH) / 128, 0);
    cell(Gbuf, cD, hD16, DHH, nullptr, 0);
    // y_t = h @ W_dec.T + b_dec  -> out[:, t, :]   (N padded 123 -> 128)
    gemm(hD16, DHH, wdec16, DHH, b_dec, nullptr, nullptr, 0, nullptr, nullptr, 0,
         out, nullptr, (long)TT * NOUT, (long)t * NOUT, DHH, NOUT, 1, 0);
  }

  // ---- in-place MDN transform over [B,T,123] -------------------------------
  {
    int total = BB * TT;
    k_mdn<<<(total + 255) / 256, 256, 0, stream>>>(out, total);
  }
}
